// DCGIN_40338332844937
// MI455X (gfx1250) — matrix operand order repaired and checked by reading, not compiled, not run on hardware
//
#include <hip/hip_runtime.h>
#include <hip/hip_bf16.h>

#define NN 50000
#define EE 800000
#define WD 14

typedef __attribute__((ext_vector_type(16))) _Float16 v16h;
typedef __attribute__((ext_vector_type(8)))  float    v8f;

__device__ __forceinline__ float sigm(float x) { return 1.0f / (1.0f + __expf(-x)); }

// ---- WMMA fragment helpers (layouts per CDNA5 ISA §7.12.2) ----
// 16-bit A 16x32: lane L (L<16) row M=L holds K=0..7 (v[0..3]) and K=16..23 (v[4..7]);
// lane L+16 holds K=8..15 and K=24..31.  As v16h elements:
//   j in 0..7  -> K = ks*8 + j
//   j in 8..15 -> K = 16 + ks*8 + (j-8)        (ks = lane>>4)
__device__ __forceinline__ v16h a_frag_f32(const float* __restrict__ rowp, int ks) {
  v16h a;
#pragma unroll
  for (int j = 0; j < 8; ++j) a[j] = (_Float16)rowp[ks * 8 + j];
#pragma unroll
  for (int j = 0; j < 8; ++j) a[8 + j] = (_Float16)rowp[16 + ks * 8 + j];
  return a;
}
__device__ __forceinline__ v16h a_frag_sum(const float* __restrict__ p,
                                           const float* __restrict__ q, int ks) {
  v16h a;
#pragma unroll
  for (int j = 0; j < 8; ++j) a[j] = (_Float16)(p[ks * 8 + j] + q[ks * 8 + j]);
#pragma unroll
  for (int j = 0; j < 8; ++j) a[8 + j] = (_Float16)(p[16 + ks * 8 + j] + q[16 + ks * 8 + j]);
  return a;
}
__device__ __forceinline__ v16h a_frag_h(const _Float16* __restrict__ rowp, int ks) {
  v16h a;
#pragma unroll
  for (int j = 0; j < 8; ++j) a[j] = rowp[ks * 8 + j];
#pragma unroll
  for (int j = 0; j < 8; ++j) a[8 + j] = rowp[16 + ks * 8 + j];
  return a;
}
// 16-bit B 32x16: element j of lane l -> K as for A (j<8: ks*8+j; j>=8: 16+ks*8+(j-8)), N = l&15.
// Weight matrix stored W[col][k] (row-major, leading dim = ld), B[k][col] = W[col][k].
__device__ __forceinline__ v16h b_frag_h(const _Float16* __restrict__ w, int ld,
                                         int col, int kb, int ks) {
  const _Float16* p = w + (size_t)col * ld + kb;
  v16h b;
#pragma unroll
  for (int j = 0; j < 8; ++j) { b[j] = p[ks * 8 + j]; b[8 + j] = p[16 + ks * 8 + j]; }
  return b;
}
__device__ __forceinline__ v16h b_frag_f32(const float* __restrict__ w, int ld,
                                           int col, int kb, int ks) {
  const float* p = w + (size_t)col * ld + kb;
  v16h b;
#pragma unroll
  for (int j = 0; j < 8; ++j) { b[j] = (_Float16)p[ks * 8 + j]; b[8 + j] = (_Float16)p[16 + ks * 8 + j]; }
  return b;
}
__device__ __forceinline__ v8f wmma16(v16h a, v16h b, v8f c) {
  return __builtin_amdgcn_wmma_f32_16x16x32_f16(false, a, false, b, (short)0, c,
                                                false, false);
}

// ---------------- utility kernels ----------------
__global__ __launch_bounds__(256) void kzero(float* __restrict__ p, int n) {
  int i = blockIdx.x * 256 + threadIdx.x;
  if (i < n) p[i] = 0.0f;
}

__global__ __launch_bounds__(256) void kconv1(const float* __restrict__ wih,
                                              const float* __restrict__ whh,
                                              _Float16* __restrict__ wc) {
  int i = blockIdx.x * 256 + threadIdx.x;  // 128*96
  if (i < 128 * 96) {
    int gc = i / 96, k = i % 96;
    float v = (k < 64) ? wih[gc * 64 + k] : whh[gc * 32 + (k - 64)];
    wc[i] = (_Float16)v;
  }
}
__global__ __launch_bounds__(256) void kconv2(const float* __restrict__ wih,
                                              const float* __restrict__ whh,
                                              _Float16* __restrict__ wc) {
  int i = blockIdx.x * 256 + threadIdx.x;  // 128*64
  if (i < 128 * 64) {
    int gc = i / 64, k = i % 64;
    float v = (k < 32) ? wih[gc * 32 + k] : whh[gc * 32 + (k - 32)];
    wc[i] = (_Float16)v;
  }
}

__global__ __launch_bounds__(256) void kdayprep(const float* __restrict__ x,
                                                float* __restrict__ feats,
                                                float* __restrict__ agg,
                                                float* __restrict__ red, int day) {
  int i = blockIdx.x * 256 + threadIdx.x;
  if (i < NN) {
    feats[(size_t)day * NN + i] = x[i * 2 + 1];
    agg[i * 2] = 0.0f;
    agg[i * 2 + 1] = 0.0f;
  }
  if (i < 128) red[i] = 0.0f;
}

// ---------------- graph kernels ----------------
__global__ __launch_bounds__(256) void kedge1(const int* __restrict__ ei,
                                              const float* __restrict__ x,
                                              float* __restrict__ agg) {
  int e = blockIdx.x * 256 + threadIdx.x;
  if (e < EE) {
    int s = ei[e];
    int d = ei[EE + e];
    atomicAdd(&agg[(size_t)d * 2 + 0], x[(size_t)s * 2 + 0]);
    atomicAdd(&agg[(size_t)d * 2 + 1], x[(size_t)s * 2 + 1]);
  }
}

// weighted scatter-add of 32-channel rows: lane = channel
__global__ __launch_bounds__(256) void kedge2(const int* __restrict__ ei,
                                              const float* __restrict__ ea,
                                              const float* __restrict__ x1,
                                              float* __restrict__ agg2) {
  int e = blockIdx.x * 8 + (threadIdx.x >> 5);
  int d = threadIdx.x & 31;
  if (e < EE) {
    int s = ei[e];
    int dn = ei[EE + e];
    float v = ea[e] * x1[(size_t)s * 32 + d];
    atomicAdd(&agg2[(size_t)dn * 32 + d], v);
  }
}

// y1 = (agg + x) @ w1^T + b1  (K=2, scalar) + fused sum/sumsq reduction
__global__ __launch_bounds__(256) void ky1(const float* __restrict__ x,
                                           const float* __restrict__ agg,
                                           const float* __restrict__ w1,
                                           const float* __restrict__ b1,
                                           float* __restrict__ y,
                                           float* __restrict__ red) {
  __shared__ float ls[64];
  int tid = threadIdx.x;
  if (tid < 64) ls[tid] = 0.0f;
  __syncthreads();
  int d = tid & 31;
  float w0 = w1[d * 2 + 0], w1v = w1[d * 2 + 1], bb = b1[d];
  int rg = blockIdx.x * 8 + (tid >> 5);  // row-group of 8 rows
  int nbase = rg * 8;
  float s = 0.0f, s2 = 0.0f;
#pragma unroll
  for (int i = 0; i < 8; ++i) {
    int n = nbase + i;
    if (n < NN) {
      float a0 = agg[(size_t)n * 2 + 0] + x[(size_t)n * 2 + 0];
      float a1 = agg[(size_t)n * 2 + 1] + x[(size_t)n * 2 + 1];
      float v = a0 * w0 + a1 * w1v + bb;
      y[(size_t)n * 32 + d] = v;
      s += v;
      s2 += v * v;
    }
  }
  atomicAdd(&ls[d], s);
  atomicAdd(&ls[32 + d], s2);
  __syncthreads();
  if (tid < 64) atomicAdd(&red[tid], ls[tid]);
}

// graph-norm stats -> per-channel scale/shift
__global__ void kstats(const float* __restrict__ red, const float* __restrict__ gw,
                       const float* __restrict__ gb, const float* __restrict__ gms,
                       float* __restrict__ ss) {
  int d = threadIdx.x;  // 32 threads
  float mu = red[d] * (1.0f / NN);
  float m2 = red[32 + d] * (1.0f / NN);
  float cc = gms[d] * mu;
  float var = m2 - 2.0f * cc * mu + cc * cc;
  float sc = gw[d] * rsqrtf(var + 1e-5f);
  ss[d] = sc;
  ss[32 + d] = gb[d] - cc * sc;
}

__global__ __launch_bounds__(256) void knorm(const float* __restrict__ y,
                                             const float* __restrict__ ss,
                                             float* __restrict__ xo) {
  int i = blockIdx.x * 256 + threadIdx.x;
  if (i < NN * 32) {
    int d = i & 31;
    xo[i] = fmaxf(y[i] * ss[d] + ss[32 + d], 0.0f);
  }
}

// y2 = (agg2 + x1) @ w2^T + b2 via WMMA (K=32) + fused GN reduction
__global__ __launch_bounds__(64) void ky2(const float* __restrict__ agg2,
                                          const float* __restrict__ x1,
                                          const float* __restrict__ w2,
                                          const float* __restrict__ b2,
                                          float* __restrict__ y,
                                          float* __restrict__ red) {
  int tid = threadIdx.x;
  int w = tid >> 5, lane = tid & 31;
  int m = lane & 15, ks = lane >> 4;
  int r0 = blockIdx.x * 16;
  int row = r0 + m;
  v16h a = a_frag_sum(agg2 + (size_t)row * 32, x1 + (size_t)row * 32, ks);
  v16h b = b_frag_f32(w2, 32, w * 16 + m, 0, ks);
  v8f c = {};
  c = wmma16(a, b, c);
  int col = w * 16 + m;
  float bb = b2[col];
  float s = 0.0f, s2 = 0.0f;
#pragma unroll
  for (int r = 0; r < 8; ++r) {
    float v = c[r] + bb;
    y[(size_t)(r0 + r + 8 * ks) * 32 + col] = v;
    s += v;
    s2 += v * v;
  }
  s += __shfl_xor(s, 16, 32);
  s2 += __shfl_xor(s2, 16, 32);
  if (ks == 0) {
    atomicAdd(&red[col], s);
    atomicAdd(&red[32 + col], s2);
  }
}

// Fused dual-LSTM: one block = 16 nodes; 4 waves, wave w owns gate columns 32w..32w+31.
__global__ __launch_bounds__(128) void klstm(
    const float* __restrict__ x1, const float* __restrict__ x2,
    float* __restrict__ h1, float* __restrict__ c1, float* __restrict__ h2,
    float* __restrict__ c2, const _Float16* __restrict__ wc1,
    const _Float16* __restrict__ wc2, const float* __restrict__ bih1,
    const float* __restrict__ bhh1, const float* __restrict__ bih2,
    const float* __restrict__ bhh2) {
  __shared__ float gls[16][128];
  __shared__ _Float16 a2[16][64];
  const int tid = threadIdx.x;
  const int w = tid >> 5;
  const int lane = tid & 31;
  const int m = lane & 15;
  const int ks = lane >> 4;
  const int r0 = blockIdx.x * 16;
  const int row = r0 + m;

  // ---- cell 1: g = [x1 | x2 | h1] @ wc1^T ----
  v8f acc0 = {}, acc1 = {};
  {
    v16h a = a_frag_f32(x1 + (size_t)row * 32, ks);
    acc0 = wmma16(a, b_frag_h(wc1, 96, w * 32 + 0 + m, 0, ks), acc0);
    acc1 = wmma16(a, b_frag_h(wc1, 96, w * 32 + 16 + m, 0, ks), acc1);
    a = a_frag_f32(x2 + (size_t)row * 32, ks);
    acc0 = wmma16(a, b_frag_h(wc1, 96, w * 32 + 0 + m, 32, ks), acc0);
    acc1 = wmma16(a, b_frag_h(wc1, 96, w * 32 + 16 + m, 32, ks), acc1);
    a = a_frag_f32(h1 + (size_t)row * 32, ks);
    acc0 = wmma16(a, b_frag_h(wc1, 96, w * 32 + 0 + m, 64, ks), acc0);
    acc1 = wmma16(a, b_frag_h(wc1, 96, w * 32 + 16 + m, 64, ks), acc1);
  }
#pragma unroll
  for (int r = 0; r < 8; ++r) {
    gls[r + 8 * ks][w * 32 + m] = acc0[r];
    gls[r + 8 * ks][w * 32 + 16 + m] = acc1[r];
  }
  __syncthreads();
#pragma unroll
  for (int it = 0; it < 4; ++it) {
    int e = tid + it * 128;  // 0..511 -> 16 rows x 32 ch
    int rr = e >> 5;
    int ch = e & 31;
    int node = r0 + rr;
    float gi = gls[rr][ch] + bih1[ch] + bhh1[ch];
    float gf = gls[rr][32 + ch] + bih1[32 + ch] + bhh1[32 + ch];
    float gg = gls[rr][64 + ch] + bih1[64 + ch] + bhh1[64 + ch];
    float go = gls[rr][96 + ch] + bih1[96 + ch] + bhh1[96 + ch];
    float cp = c1[(size_t)node * 32 + ch];
    float cn = sigm(gf) * cp + sigm(gi) * tanhf(gg);
    float hn = sigm(go) * tanhf(cn);
    c1[(size_t)node * 32 + ch] = cn;
    h1[(size_t)node * 32 + ch] = hn;
    a2[rr][ch] = (_Float16)hn;
    a2[rr][32 + ch] = (_Float16)h2[(size_t)node * 32 + ch];
  }
  __syncthreads();

  // ---- cell 2: g = [h1_new | h2] @ wc2^T ----
  v8f d0 = {}, d1 = {};
  {
    v16h a = a_frag_h(&a2[m][0], ks);
    d0 = wmma16(a, b_frag_h(wc2, 64, w * 32 + 0 + m, 0, ks), d0);
    d1 = wmma16(a, b_frag_h(wc2, 64, w * 32 + 16 + m, 0, ks), d1);
    a = a_frag_h(&a2[m][32], ks);
    d0 = wmma16(a, b_frag_h(wc2, 64, w * 32 + 0 + m, 32, ks), d0);
    d1 = wmma16(a, b_frag_h(wc2, 64, w * 32 + 16 + m, 32, ks), d1);
  }
#pragma unroll
  for (int r = 0; r < 8; ++r) {
    gls[r + 8 * ks][w * 32 + m] = d0[r];
    gls[r + 8 * ks][w * 32 + 16 + m] = d1[r];
  }
  __syncthreads();
#pragma unroll
  for (int it = 0; it < 4; ++it) {
    int e = tid + it * 128;
    int rr = e >> 5;
    int ch = e & 31;
    int node = r0 + rr;
    float gi = gls[rr][ch] + bih2[ch] + bhh2[ch];
    float gf = gls[rr][32 + ch] + bih2[32 + ch] + bhh2[32 + ch];
    float gg = gls[rr][64 + ch] + bih2[64 + ch] + bhh2[64 + ch];
    float go = gls[rr][96 + ch] + bih2[96 + ch] + bhh2[96 + ch];
    float cp = c2[(size_t)node * 32 + ch];
    float cn = sigm(gf) * cp + sigm(gi) * tanhf(gg);
    float hn = sigm(go) * tanhf(cn);
    c2[(size_t)node * 32 + ch] = cn;
    h2[(size_t)node * 32 + ch] = hn;
  }
}

__global__ __launch_bounds__(256) void kfinal(const float* __restrict__ feats,
                                              const float* __restrict__ h1,
                                              const float* __restrict__ h2,
                                              const float* __restrict__ lin_w,
                                              const float* __restrict__ lin_b,
                                              float* __restrict__ out) {
  int n = blockIdx.x * 256 + threadIdx.x;
  if (n < NN) {
    float acc = lin_b[0];
#pragma unroll
    for (int j = 0; j < WD; ++j)
      acc += fmaxf(feats[(size_t)j * NN + n], 0.0f) * lin_w[j];
    const float* h1p = h1 + (size_t)n * 32;
    const float* h2p = h2 + (size_t)n * 32;
#pragma unroll
    for (int d = 0; d < 32; ++d) {
      acc += fmaxf(h1p[d], 0.0f) * lin_w[WD + d];
      acc += fmaxf(h2p[d], 0.0f) * lin_w[WD + 32 + d];
    }
    out[n] = acc;
  }
}

extern "C" void kernel_launch(void* const* d_in, const int* in_sizes, int n_in,
                              void* d_out, int out_size, void* d_ws, size_t ws_size,
                              hipStream_t stream) {
  const float* x_days = (const float*)d_in[0];
  const int* eidx = (const int*)d_in[1];
  const float* eattr = (const float*)d_in[2];
  const float* w1 = (const float*)d_in[3];
  const float* b1 = (const float*)d_in[4];
  const float* w2 = (const float*)d_in[5];
  const float* b2 = (const float*)d_in[6];
  const float* gn1_w = (const float*)d_in[7];
  const float* gn1_b = (const float*)d_in[8];
  const float* gn1_ms = (const float*)d_in[9];
  const float* gn2_w = (const float*)d_in[10];
  const float* gn2_b = (const float*)d_in[11];
  const float* gn2_ms = (const float*)d_in[12];
  const float* l1_wih = (const float*)d_in[13];
  const float* l1_whh = (const float*)d_in[14];
  const float* l1_bih = (const float*)d_in[15];
  const float* l1_bhh = (const float*)d_in[16];
  const float* l2_wih = (const float*)d_in[17];
  const float* l2_whh = (const float*)d_in[18];
  const float* l2_bih = (const float*)d_in[19];
  const float* l2_bhh = (const float*)d_in[20];
  const float* lin_w = (const float*)d_in[21];
  const float* lin_b = (const float*)d_in[22];

  float* out = (float*)d_out;
  float* h1 = out + NN;
  float* c1 = h1 + (size_t)NN * 32;
  float* h2 = c1 + (size_t)NN * 32;
  float* c2 = h2 + (size_t)NN * 32;

  float* ws = (float*)d_ws;
  float* agg = ws;                       // N*2
  float* ybuf = agg + (size_t)NN * 2;    // N*32
  float* agg2 = ybuf + (size_t)NN * 32;  // N*32
  float* x1b = agg2 + (size_t)NN * 32;   // N*32
  float* x2b = x1b + (size_t)NN * 32;    // N*32
  float* feats = x2b + (size_t)NN * 32;  // N*WD
  float* red = feats + (size_t)NN * WD;  // 128 (red1|red2)
  float* ss1 = red + 128;                // 64
  float* ss2 = ss1 + 64;                 // 64
  _Float16* wc1 = (_Float16*)(ss2 + 64); // 128*96 halfs
  _Float16* wc2 = wc1 + 128 * 96;        // 128*64 halfs

  // one-time prep: f16 weight concat + zero h/c state
  kconv1<<<48, 256, 0, stream>>>(l1_wih, l1_whh, wc1);
  kconv2<<<32, 256, 0, stream>>>(l2_wih, l2_whh, wc2);
  kzero<<<25000, 256, 0, stream>>>(h1, NN * 128);

  for (int day = 0; day < WD; ++day) {
    const float* x = x_days + (size_t)day * NN * 2;
    const int* ei = eidx + (size_t)day * 2 * EE;
    const float* ea = eattr + (size_t)day * EE;

    kdayprep<<<196, 256, 0, stream>>>(x, feats, agg, red, day);
    kzero<<<6250, 256, 0, stream>>>(agg2, NN * 32);
    kedge1<<<3125, 256, 0, stream>>>(ei, x, agg);
    ky1<<<782, 256, 0, stream>>>(x, agg, w1, b1, ybuf, red);
    kstats<<<1, 32, 0, stream>>>(red, gn1_w, gn1_b, gn1_ms, ss1);
    knorm<<<6250, 256, 0, stream>>>(ybuf, ss1, x1b);
    kedge2<<<100000, 256, 0, stream>>>(ei, ea, x1b, agg2);
    ky2<<<3125, 64, 0, stream>>>(agg2, x1b, w2, b2, ybuf, red + 64);
    kstats<<<1, 32, 0, stream>>>(red + 64, gn2_w, gn2_b, gn2_ms, ss2);
    knorm<<<6250, 256, 0, stream>>>(ybuf, ss2, x2b);
    klstm<<<3125, 128, 0, stream>>>(x1b, x2b, h1, c1, h2, c2, wc1, wc2, l1_bih,
                                    l1_bhh, l2_bih, l2_bhh);
  }
  kfinal<<<196, 256, 0, stream>>>(feats, h1, h2, lin_w, lin_b, out);
}
